// MultiHeadedSelfAttentionWithRelPos_7310034338302
// MI455X (gfx1250) — matrix operand order repaired and checked by reading, not compiled, not run on hardware
//
#include <hip/hip_runtime.h>
#include <stdint.h>

// CDNA5 / gfx1250, wave32. f16-input WMMA with fp32 accumulation.
// Roofline: ~70 GFLOP total; mandatory 256MB weights write => ~14us memory
// floor at 23.3 TB/s. f16 WMMA (16x16x32) keeps matrix math (~28us) roughly
// balanced against that floor; fp32 WMMA (16x16x4) would be ~8x slower.

typedef __attribute__((ext_vector_type(16))) _Float16 v16h;
typedef __attribute__((ext_vector_type(8)))  _Float16 v8h;
typedef __attribute__((ext_vector_type(4)))  _Float16 v4h;
typedef __attribute__((ext_vector_type(8)))  float    v8f;
typedef __attribute__((ext_vector_type(4)))  int      v4i;

#define BATCH  4
#define TT     1024
#define DMODEL 1024
#define NHEAD  16
#define DKH    64
#define RROWS  (BATCH * TT)
#define SCLD   (TT + 8)   // padded LDS row stride

#if defined(__HIP_DEVICE_COMPILE__) && __has_builtin(__builtin_amdgcn_global_load_async_to_lds_b128)
#define HAS_ASYNC_LDS 1
#else
#define HAS_ASYNC_LDS 0
#endif

union H16 { v16h v; v8h h[2]; };

// ---- 16-bit WMMA operand layouts (CDNA5 ISA 7.12.2) ----
// A (16x32): lane m=lane&15, kh=lane>>4; lane holds K in {8kh..8kh+7, 16+8kh..16+8kh+7},
//            sequential in the fragment -> contiguous in a row-major f16 tile.
// B (32x16): lane n=lane&15, kh=lane>>4; lane holds K in {8kh..8kh+7, 16+8kh..16+8kh+7},
//            sequential in the fragment (same order as A).
__device__ __forceinline__ int kb_base(int j, int kh) { return (j & 3) * 2 + (j >> 2) * 16 + kh * 8; }

__device__ __forceinline__ v8f wmma16(v16h a, v16h b, v8f c) {
  return __builtin_amdgcn_wmma_f32_16x16x32_f16(false, a, false, b, (short)0, c,
                                                false, false);
}

__device__ __forceinline__ float wave_max(float v) {
#pragma unroll
  for (int o = 16; o; o >>= 1) v = fmaxf(v, __shfl_xor(v, o, 32));
  return v;
}
__device__ __forceinline__ float wave_sum(float v) {
#pragma unroll
  for (int o = 16; o; o >>= 1) v += __shfl_xor(v, o, 32);
  return v;
}

#if HAS_ASYNC_LDS
// 16B global -> LDS async copy (GLOBAL_LOAD_ASYNC_TO_LDS_B128, ASYNCcnt).
// Builtin signature (from hipcc diagnostic): (AS1 int4*, AS3 int4*, imm, imm).
__device__ __forceinline__ void async_ld16(const float* g, float* l) {
  __builtin_amdgcn_global_load_async_to_lds_b128(
      (__attribute__((address_space(1))) v4i*)(uintptr_t)g,
      (__attribute__((address_space(3))) v4i*)(unsigned)(uintptr_t)l,
      0, 0);
}
#endif

// ============================================================================
// Generic GEMM: C[M,N] = A[M,K] @ B[K,N] (+ bias[N]).  f32 in/out, f16 WMMA.
// Block tile 64x128, 8 waves of 32x32, K-step 32.  Software-pipelined with
// double-buffered f16 LDS tiles: chunk i+1's global loads issue before chunk
// i's WMMAs, so HBM latency overlaps matrix math.  A row-major, B transposed
// so all fragment reads are contiguous ds_load_b128.
// ============================================================================
__global__ __launch_bounds__(256) void gemm_rcr(const float* __restrict__ A,
                                                const float* __restrict__ Bw,
                                                const float* __restrict__ bias,
                                                float* __restrict__ C,
                                                int M, int N, int K) {
  __shared__ _Float16 As[2][64][40];    // 64x32 tiles, +8 pad (row=80B, 16B-aligned)
  __shared__ _Float16 Bt[2][128][40];   // transposed 32x128 tiles: Bt[buf][n][k]

  const int bm   = blockIdx.y * 64;
  const int bn   = blockIdx.x * 128;
  const int tid  = threadIdx.x;
  const int wave = tid >> 5;
  const int lane = tid & 31;
  const int lrow = lane & 15, kh = lane >> 4;
  const int wm   = (wave >> 2) * 32;  // 0 / 32
  const int wn   = (wave & 3) * 32;   // 0..96

  // staging ownership: A -> row ar, cols ac4..+7; B -> rows br0..+3, cols bc4..+3
  const int ar  = tid >> 2,  ac4 = (tid & 3) * 8;
  const int br0 = wave * 4,  bc4 = lane * 4;

  v8f acc[2][2] = {{{}, {}}, {{}, {}}};
  float4 ra0, ra1, rb[4];

  // issue global loads for one K-chunk into registers (batched, no waits between)
  auto gload = [&](int k0) {
    const float* ap = &A[(size_t)(bm + ar) * K + k0 + ac4];
    ra0 = *(const float4*)(ap);
    ra1 = *(const float4*)(ap + 4);
#pragma unroll
    for (int p = 0; p < 4; ++p)
      rb[p] = *(const float4*)&Bw[(size_t)(k0 + br0 + p) * N + bn + bc4];
  };
  // convert staged registers to f16 and store to LDS buffer `buf`
  auto lstore = [&](int buf) {
    v8h h;
    h[0] = (_Float16)ra0.x; h[1] = (_Float16)ra0.y;
    h[2] = (_Float16)ra0.z; h[3] = (_Float16)ra0.w;
    h[4] = (_Float16)ra1.x; h[5] = (_Float16)ra1.y;
    h[6] = (_Float16)ra1.z; h[7] = (_Float16)ra1.w;
    *(v8h*)&As[buf][ar][ac4] = h;
#pragma unroll
    for (int i = 0; i < 4; ++i) {   // column bc4+i gets 4 contiguous k-halves
      v4h hc;
      hc[0] = (_Float16)((const float*)&rb[0])[i];
      hc[1] = (_Float16)((const float*)&rb[1])[i];
      hc[2] = (_Float16)((const float*)&rb[2])[i];
      hc[3] = (_Float16)((const float*)&rb[3])[i];
      *(v4h*)&Bt[buf][bc4 + i][br0] = hc;
    }
  };

  const int steps = K >> 5;
  gload(0);
  lstore(0);
  __syncthreads();

  for (int i = 0; i < steps; ++i) {
    const int cur = i & 1;
    if (i + 1 < steps) gload((i + 1) << 5);   // overlap next chunk with WMMAs

    H16 af[2], bf[2];
#pragma unroll
    for (int mi = 0; mi < 2; ++mi) {
      af[mi].h[0] = *(const v8h*)&As[cur][wm + mi * 16 + lrow][8 * kh];
      af[mi].h[1] = *(const v8h*)&As[cur][wm + mi * 16 + lrow][16 + 8 * kh];
    }
#pragma unroll
    for (int ni = 0; ni < 2; ++ni) {
      bf[ni].h[0] = *(const v8h*)&Bt[cur][wn + ni * 16 + lrow][8 * kh];
      bf[ni].h[1] = *(const v8h*)&Bt[cur][wn + ni * 16 + lrow][16 + 8 * kh];
    }
#pragma unroll
    for (int mi = 0; mi < 2; ++mi)
#pragma unroll
      for (int ni = 0; ni < 2; ++ni)
        acc[mi][ni] = wmma16(af[mi].v, bf[ni].v, acc[mi][ni]);

    if (i + 1 < steps) {
      lstore(1 - cur);        // writes the buffer not being read this iteration
      __syncthreads();
    }
  }

  // epilogue: C/D layout — lane col=lane&15, vgpr r -> row r + 8*(lane>>4)
#pragma unroll
  for (int mi = 0; mi < 2; ++mi) {
#pragma unroll
    for (int ni = 0; ni < 2; ++ni) {
      int col  = bn + wn + ni * 16 + lrow;
      float bv = bias ? bias[col] : 0.0f;
#pragma unroll
      for (int r = 0; r < 8; ++r) {
        int row = bm + wm + mi * 16 + kh * 8 + r;
        C[(size_t)row * N + col] = acc[mi][ni][r] + bv;
      }
    }
  }
}

// ============================================================================
// Fused rel-pos attention.  One block per (b, h, 16-row q-tile); 8 waves.
//   phase 1: AC=(q+u)k^T, BDraw=(q+v)p^T via WMMA (rows q..q+16) -> LDS;
//            k/p tiles staged via GLOBAL_LOAD_ASYNC_TO_LDS_B128 when available
//   phase 2: TXL shift + mask + softmax (wave32 shuffles); weights -> d_out
//            (f32) and -> LDS (f16)
//   phase 3: ctx(16x64) = weights @ V via WMMA -> ws
// ============================================================================
__global__ __launch_bounds__(256) void attn_relpos(
    const float* __restrict__ qkv,    // [B,T,3,NH,DK]
    const float* __restrict__ pmat,   // [B,T,NH,DK]
    const float* __restrict__ posu,   // [NH,DK]
    const float* __restrict__ posv,   // [NH,DK]
    const unsigned char* __restrict__ mask,  // [B,1,T] bool
    float* __restrict__ wts,          // [B,NH,T,T]
    float* __restrict__ ctx)          // [B,T,DM]
{
  extern __shared__ char smem[];
  float*    ac_s  = (float*)smem;                  // [16][SCLD]
  float*    bd_s  = ac_s + 16 * SCLD;              // [17][SCLD] (extra row q+16)
  _Float16* wh_s  = (_Float16*)(bd_s + 17 * SCLD); // [16][SCLD]
  float*    stage = (float*)(wh_s + 16 * SCLD);    // 8 waves x 2 x [16][64] f32

  const int qg = blockIdx.x, h = blockIdx.y, b = blockIdx.z;
  const int qbase = qg * 16;
  const int tid = threadIdx.x, wave = tid >> 5, lane = tid & 31;
  const int lrow = lane & 15, kh = lane >> 4;

  const size_t rq = (size_t)3 * DMODEL;            // qkv row stride
  const float* qp = qkv + (size_t)b * TT * rq + 0 * DMODEL + h * DKH;
  const float* kp = qkv + (size_t)b * TT * rq + 1 * DMODEL + h * DKH;
  const float* vp = qkv + (size_t)b * TT * rq + 2 * DMODEL + h * DKH;
  const float* pp = pmat + (size_t)b * TT * DMODEL + h * DKH;
  const float* pu = posu + h * DKH;
  const float* pv = posv + h * DKH;

  // ---- A fragments: qu rows q.., qv rows q.., qv rows q+1.. (rel-shift row)
  v16h aqu[2], aqv[2], aqv2[2];
  {
    int r2 = qbase + lrow + 1; if (r2 > TT - 1) r2 = TT - 1;  // clamp, value unused
#pragma unroll
    for (int ks = 0; ks < 2; ++ks) {
#pragma unroll
      for (int j = 0; j < 8; ++j) {
        int kk = ((j & 3) * 2) + ((j >> 2) * 16) + kh * 8 + ks * 32;  // A layout
        float2 q0 = *(const float2*)&qp[(size_t)(qbase + lrow) * rq + kk];
        float2 q1 = *(const float2*)&qp[(size_t)r2 * rq + kk];
        aqu[ks][2 * j]      = (_Float16)(q0.x + pu[kk]);
        aqu[ks][2 * j + 1]  = (_Float16)(q0.y + pu[kk + 1]);
        aqv[ks][2 * j]      = (_Float16)(q0.x + pv[kk]);
        aqv[ks][2 * j + 1]  = (_Float16)(q0.y + pv[kk + 1]);
        aqv2[ks][2 * j]     = (_Float16)(q1.x + pv[kk]);
        aqv2[ks][2 * j + 1] = (_Float16)(q1.y + pv[kk + 1]);
      }
    }
  }

#if HAS_ASYNC_LDS
  float* kst = stage + wave * (2 * 16 * DKH);
  float* pst = kst + 16 * DKH;
#else
  (void)stage;
#endif

  // ---- phase 1: score tiles; wave handles key tiles kt, kt+8, ...
  for (int kt = wave; kt < TT / 16; kt += 8) {
    const int kb0 = kt * 16;
#if HAS_ASYNC_LDS
    {
      // drain prior LDS reads of the staging buffer, then async-copy 2x4KB
      asm volatile("s_wait_dscnt 0x0" ::: "memory");
      const float* gk = kp + (size_t)kb0 * rq;
      const float* gq = pp + (size_t)kb0 * DMODEL;
#pragma unroll
      for (int i = 0; i < 8; ++i) {
        int idx = i * 128 + lane * 4;          // float index within [16][64]
        int row = idx >> 6, col = idx & 63;
        async_ld16(gk + (size_t)row * rq + col, kst + idx);
        async_ld16(gq + (size_t)row * DMODEL + col, pst + idx);
      }
      asm volatile("s_wait_asynccnt 0x0" ::: "memory");
    }
#endif
    v8f acw = {}, bd = {}, bd2 = {};
#pragma unroll
    for (int ks = 0; ks < 2; ++ks) {
      v16h bk, bp;
#pragma unroll
      for (int j = 0; j < 8; ++j) {
        int kk = kb_base(j, kh) + ks * 32;
#if HAS_ASYNC_LDS
        float2 kv = *(const float2*)&kst[lrow * DKH + kk];
        float2 pz = *(const float2*)&pst[lrow * DKH + kk];
#else
        float2 kv = *(const float2*)&kp[(size_t)(kb0 + lrow) * rq + kk];
        float2 pz = *(const float2*)&pp[(size_t)(kb0 + lrow) * DMODEL + kk];
#endif
        bk[2 * j] = (_Float16)kv.x; bk[2 * j + 1] = (_Float16)kv.y;
        bp[2 * j] = (_Float16)pz.x; bp[2 * j + 1] = (_Float16)pz.y;
      }
      acw = wmma16(aqu[ks],  bk, acw);
      bd  = wmma16(aqv[ks],  bp, bd);
      bd2 = wmma16(aqv2[ks], bp, bd2);
    }
#pragma unroll
    for (int r = 0; r < 8; ++r) {
      int rr = r + kh * 8;
      ac_s[rr * SCLD + kb0 + lrow] = acw[r];
      bd_s[rr * SCLD + kb0 + lrow] = bd[r];
    }
    if (kh == 1) bd_s[16 * SCLD + kb0 + lrow] = bd2[7];  // bdraw row q+16
  }
  __syncthreads();

  // ---- phase 2: shift + mask + softmax; 2 rows per wave
  const unsigned char* mrow = mask + (size_t)b * TT;
#pragma unroll
  for (int rr = 0; rr < 2; ++rr) {
    const int r  = wave * 2 + rr;
    const int qi = qbase + r;
    float mx = -3.4e38f;
    for (int c = lane; c < TT; c += 32) {
      float bdv;
      if (c <= qi)          bdv = bd_s[r * SCLD + (c - qi + TT - 1)];
      else if (c == qi + 1) bdv = 0.0f;
      else                  bdv = bd_s[(r + 1) * SCLD + (c - qi - 2)];
      float s = (ac_s[r * SCLD + c] + bdv) * 0.125f;  // 1/sqrt(64)
      if (!mrow[c]) s = -INFINITY;
      ac_s[r * SCLD + c] = s;
      mx = fmaxf(mx, s);
    }
    mx = wave_max(mx);
    float sum = 0.0f;
    for (int c = lane; c < TT; c += 32) {
      float e = __expf(ac_s[r * SCLD + c] - mx);
      ac_s[r * SCLD + c] = e;
      sum += e;
    }
    sum = wave_sum(sum);
    const float rinv = 1.0f / sum;
    float* wg = wts + (((size_t)(b * NHEAD + h) * TT + qi) * TT);
    for (int c = lane; c < TT; c += 32) {
      float w = ac_s[r * SCLD + c] * rinv;
      wg[c] = w;
      wh_s[r * SCLD + c] = (_Float16)w;
    }
  }
  __syncthreads();

  // ---- phase 3: ctx(16x64) = weights(16x1024) @ V(1024x64); waves 0..3
  if (wave < 4) {
    const int dcol = wave * 16;
    v8f accp = {};
    for (int kc = 0; kc < TT; kc += 32) {
      H16 aw;   // weights already f16 in LDS; A-layout is contiguous
      aw.h[0] = *(const v8h*)&wh_s[lrow * SCLD + kc + 8 * kh];
      aw.h[1] = *(const v8h*)&wh_s[lrow * SCLD + kc + 16 + 8 * kh];
      v16h bv;
#pragma unroll
      for (int j = 0; j < 8; ++j) {
        int kk = kb_base(j, kh);
        bv[2 * j]     = (_Float16)vp[(size_t)(kc + kk)     * rq + dcol + lrow];
        bv[2 * j + 1] = (_Float16)vp[(size_t)(kc + kk + 1) * rq + dcol + lrow];
      }
      accp = wmma16(aw.v, bv, accp);
    }
    float* cb = ctx + ((size_t)b * TT + qbase) * DMODEL + h * DKH + dcol;
#pragma unroll
    for (int r = 0; r < 8; ++r)
      cb[(size_t)(r + kh * 8) * DMODEL + lrow] = accp[r];
  }
}

// ============================================================================
extern "C" void kernel_launch(void* const* d_in, const int* in_sizes, int n_in,
                              void* d_out, int out_size, void* d_ws, size_t ws_size,
                              hipStream_t stream) {
  const float* x    = (const float*)d_in[0];
  const unsigned char* mask = (const unsigned char*)d_in[1];
  const float* pos  = (const float*)d_in[2];
  const float* Wqkv = (const float*)d_in[3];
  const float* bqkv = (const float*)d_in[4];
  const float* Wpos = (const float*)d_in[5];
  const float* posu = (const float*)d_in[6];
  const float* posv = (const float*)d_in[7];
  const float* Wout = (const float*)d_in[8];
  const float* bout = (const float*)d_in[9];

  float* out = (float*)d_out;                     // [ctx (R*DM) | weights]
  float* qkv = (float*)d_ws;                      // R*3*DM f32
  float* p   = qkv + (size_t)RROWS * 3 * DMODEL;  // R*DM f32
  float* ctx = p   + (size_t)RROWS * DMODEL;      // R*DM f32

  dim3 blk(256);

  // 1) qkv = x @ Wqkv + bqkv
  gemm_rcr<<<dim3(3 * DMODEL / 128, RROWS / 64), blk, 0, stream>>>(
      x, Wqkv, bqkv, qkv, RROWS, 3 * DMODEL, DMODEL);

  // 2) p = pos @ Wpos
  gemm_rcr<<<dim3(DMODEL / 128, RROWS / 64), blk, 0, stream>>>(
      pos, Wpos, nullptr, p, RROWS, DMODEL, DMODEL);

  // 3) fused attention (scores, shift, softmax, weights out, weights@V)
  //    LDS: 33*SCLD f32 + 16*SCLD f16 + 8*2*16*64 f32 async staging (~230 KB,
  //    within the 320 KB/WGP budget; always reserved so host/device agree).
  size_t shmem = (size_t)(33 * SCLD) * sizeof(float) + (size_t)(16 * SCLD) * 2 +
                 (size_t)(8 * 2 * 16 * DKH) * sizeof(float);
  attn_relpos<<<dim3(TT / 16, NHEAD, BATCH), blk, shmem, stream>>>(
      qkv, p, posu, posv, mask, out + (size_t)RROWS * DMODEL, ctx);

  // 4) out = ctx @ Wout + bout
  gemm_rcr<<<dim3(DMODEL / 128, RROWS / 64), blk, 0, stream>>>(
      ctx, Wout, bout, out, RROWS, DMODEL, DMODEL);
}
